// EdgeMLP_alt_74131135529469
// MI455X (gfx1250) — matrix operand order repaired and checked by reading, not compiled, not run on hardware
//
#include <hip/hip_runtime.h>
#include <hip/hip_bf16.h>

// ---------------------------------------------------------------------------
// EdgeMLP fused kernel for MI455X (gfx1250, wave32, WMMA f16->f32).
//
// Reference: x = [h[src] || h[dst]]  (E x 256)
//   x = BN0(x) @ W0^T + b0                    -> (E x 128)
//   x = BN1(relu(x)) @ W1^T + b1              -> (E x 128)
//   x = BN2(relu(x)) @ W2^T + b2              -> (E x 1)
//
// BN folded into weights: bn(x)@W^T = x@(W*s)^T + (b + t@W^T).
// Whole pipeline fused; edge tile lives in LDS; h pre-converted to fp16.
// Persistent blocks amortize register-resident weight B-fragments.
// A-fragments are loaded 4-at-a-time per K-step so the ds_load latency of
// fragments 2..4 hides under the preceding WMMAs (staggered s_wait_dscnt).
// ---------------------------------------------------------------------------

typedef __attribute__((ext_vector_type(16))) _Float16 v16h;
typedef __attribute__((ext_vector_type(8)))  _Float16 v8h;
typedef __attribute__((ext_vector_type(8)))  float    v8f;

#define BN_EPS 1e-5f
#define TILE_E 64
#define SX 264   // lds_x row stride in halves (256 + 8 pad)
#define SY 136   // lds_y row stride in halves (128 + 8 pad)

// --------------------------- prep: fold BN into weights --------------------
__global__ __launch_bounds__(256) void fold_kernel(
    const float* __restrict__ W0, const float* __restrict__ b0,
    const float* __restrict__ W1, const float* __restrict__ b1,
    const float* __restrict__ W2, const float* __restrict__ b2,
    const float* __restrict__ g0, const float* __restrict__ be0,
    const float* __restrict__ g1, const float* __restrict__ be1,
    const float* __restrict__ g2, const float* __restrict__ be2,
    const float* __restrict__ rm0, const float* __restrict__ rv0,
    const float* __restrict__ rm1, const float* __restrict__ rv1,
    const float* __restrict__ rm2, const float* __restrict__ rv2,
    _Float16* __restrict__ W0f, _Float16* __restrict__ W1f,
    _Float16* __restrict__ w2f,
    float* __restrict__ b0f, float* __restrict__ b1f, float* __restrict__ b2f)
{
    const int t = threadIdx.x;
    for (int idx = t; idx < 128 * 256; idx += 256) {
        int k = idx & 255;
        float s = g0[k] * rsqrtf(rv0[k] + BN_EPS);
        W0f[idx] = (_Float16)(W0[idx] * s);
    }
    for (int idx = t; idx < 128 * 128; idx += 256) {
        int k = idx & 127;
        float s = g1[k] * rsqrtf(rv1[k] + BN_EPS);
        W1f[idx] = (_Float16)(W1[idx] * s);
    }
    if (t < 128) {
        float s2 = g2[t] * rsqrtf(rv2[t] + BN_EPS);
        w2f[t] = (_Float16)(W2[t] * s2);

        float acc0 = b0[t];
        for (int k = 0; k < 256; ++k) {
            float s = g0[k] * rsqrtf(rv0[k] + BN_EPS);
            float tt = be0[k] - rm0[k] * s;
            acc0 += tt * W0[t * 256 + k];
        }
        b0f[t] = acc0;

        float acc1 = b1[t];
        for (int k = 0; k < 128; ++k) {
            float s = g1[k] * rsqrtf(rv1[k] + BN_EPS);
            float tt = be1[k] - rm1[k] * s;
            acc1 += tt * W1[t * 128 + k];
        }
        b1f[t] = acc1;
    }
    if (t == 0) {
        float acc = b2[0];
        for (int k = 0; k < 128; ++k) {
            float s = g2[k] * rsqrtf(rv2[k] + BN_EPS);
            float tt = be2[k] - rm2[k] * s;
            acc += tt * W2[k];
        }
        b2f[0] = acc;
    }
}

// --------------------------- prep: h -> fp16 --------------------------------
__global__ __launch_bounds__(256) void convert_h_kernel(
    const float* __restrict__ h, _Float16* __restrict__ hh, int total)
{
    int i = blockIdx.x * 256 + threadIdx.x;
    if (i < total) hh[i] = (_Float16)h[i];
}

// A fragment loader (contiguous K order matching B: lanes 0-15 hold
// K {0..15}, lanes 16-31 hold K {16..31} -> two 16B LDS loads).
__device__ __forceinline__ v16h load_a_frag(const _Float16* base, int stride,
                                            int m0, int kk, int lh, int hi)
{
    const _Float16* q = base + (m0 + lh) * stride + kk * 32 + hi * 16;
    v8h lo = *(const v8h*)(q);
    v8h hv = *(const v8h*)(q + 8);
    v16h a;
#pragma unroll
    for (int i = 0; i < 8; ++i) { a[i] = lo[i]; a[i + 8] = hv[i]; }
    return a;
}

// --------------------------- main fused kernel ------------------------------
// 256 threads = 8 wave32s.  Each block grid-strides over 64-edge tiles.
// Wave w owns output columns [16w, 16w+16); its folded-weight B-fragments
// stay in registers across ALL tiles the block processes.
__global__ __launch_bounds__(256) void edge_mlp_kernel(
    const _Float16* __restrict__ hh,           // [N][128] fp16
    const int* __restrict__ src, const int* __restrict__ dst,
    const _Float16* __restrict__ W0f,          // [128][256] folded, fp16
    const float* __restrict__ b0f,             // [128]
    const _Float16* __restrict__ W1f,          // [128][128] folded, fp16
    const float* __restrict__ b1f,             // [128]
    const _Float16* __restrict__ w2f,          // [128] folded, fp16
    const float* __restrict__ b2f,             // [1]
    float* __restrict__ out, int E, int num_tiles)
{
    __shared__ _Float16 lds_x[TILE_E * SX];    // edge tile  (E x 256) fp16
    __shared__ _Float16 lds_y[TILE_E * SY];    // stage-0 out (E x 128) fp16
    __shared__ _Float16 lds_w2[128];

    const int t    = threadIdx.x;
    const int wave = t >> 5;
    const int lane = t & 31;
    const int lh   = lane & 15;     // low 4 bits of lane
    const int hi   = lane >> 4;     // lane group (0: lanes 0-15, 1: 16-31)

    // ---- B fragments (contiguous K order as in load_a_frag: lane = col, lanes 0-15
    //      hold K=k0..k0+15 packed 2/VGPR, lanes 16-31 hold K=k0+16..k0+31).
    //      Since B = Wf^T, lane reads row Wf[j] contiguously: one 32B load.
    const int j0 = wave * 16;
    const int j  = j0 + lh;
    v16h B0[8], B1[4];
#pragma unroll
    for (int kk = 0; kk < 8; ++kk)
        B0[kk] = *(const v16h*)(W0f + j * 256 + kk * 32 + hi * 16);
#pragma unroll
    for (int kk = 0; kk < 4; ++kk)
        B1[kk] = *(const v16h*)(W1f + j * 128 + kk * 32 + hi * 16);

    const float bias0 = b0f[j];
    const float bias1 = b1f[j];
    const float bias2 = b2f[0];
    if (t < 128) lds_w2[t] = w2f[t];

    for (int tile = blockIdx.x; tile < num_tiles; tile += gridDim.x) {
        const int e0 = tile * TILE_E;

        // ---- gather: 4 threads per edge row, each copies 64 halves (128B)
        {
            const int er  = t >> 2;                 // edge row within tile
            const int seg = t & 3;                  // quarter of 256-wide row
            const int e   = e0 + er;
            const int ec  = (e < E) ? e : (E - 1);  // clamp (tail tile)
            const int node = (seg < 2) ? src[ec] : dst[ec];
            const _Float16* gp = hh + (size_t)node * 128 + (seg & 1) * 64;
            _Float16* lp = lds_x + er * SX + seg * 64;
#pragma unroll
            for (int i = 0; i < 8; ++i)
                *(v8h*)(lp + i * 8) = *(const v8h*)(gp + i * 8);
        }
        __syncthreads();

        // ---- stage 0: (64 x 256) @ W0f^T, bias+relu -> lds_y (fp16)
        //      Per K-step: load all 4 m-tile A fragments (8 ds_load_b128),
        //      then 4 WMMAs sharing B0[kk] -> staggered s_wait_dscnt.
        {
            v8f cacc[4] = {v8f{}, v8f{}, v8f{}, v8f{}};
#pragma unroll
            for (int kk = 0; kk < 8; ++kk) {
                v16h a[4];
#pragma unroll
                for (int ms = 0; ms < 4; ++ms)
                    a[ms] = load_a_frag(lds_x, SX, ms * 16, kk, lh, hi);
#pragma unroll
                for (int ms = 0; ms < 4; ++ms)
                    cacc[ms] = __builtin_amdgcn_wmma_f32_16x16x32_f16(
                            false, a[ms], false, B0[kk], (short)0, cacc[ms],
                            false, false);
            }
            // C/D layout: lane -> col n=lh, VGPR r -> row m0 + 8*hi + r
#pragma unroll
            for (int ms = 0; ms < 4; ++ms) {
#pragma unroll
                for (int r = 0; r < 8; ++r) {
                    float v = cacc[ms][r] + bias0;
                    v = v > 0.0f ? v : 0.0f;
                    lds_y[(ms * 16 + hi * 8 + r) * SY + j] = (_Float16)v;
                }
            }
        }
        __syncthreads();

        // ---- stage 1: (64 x 128) @ W1f^T, bias+relu -> y2 (reuse lds_x)
        _Float16* y2 = lds_x;
        {
            v8f cacc[4] = {v8f{}, v8f{}, v8f{}, v8f{}};
#pragma unroll
            for (int kk = 0; kk < 4; ++kk) {
                v16h a[4];
#pragma unroll
                for (int ms = 0; ms < 4; ++ms)
                    a[ms] = load_a_frag(lds_y, SY, ms * 16, kk, lh, hi);
#pragma unroll
                for (int ms = 0; ms < 4; ++ms)
                    cacc[ms] = __builtin_amdgcn_wmma_f32_16x16x32_f16(
                            false, a[ms], false, B1[kk], (short)0, cacc[ms],
                            false, false);
            }
#pragma unroll
            for (int ms = 0; ms < 4; ++ms) {
#pragma unroll
                for (int r = 0; r < 8; ++r) {
                    float v = cacc[ms][r] + bias1;
                    v = v > 0.0f ? v : 0.0f;
                    y2[(ms * 16 + hi * 8 + r) * SY + j] = (_Float16)v;
                }
            }
        }
        __syncthreads();

        // ---- stage 2: per-edge 128-dot with folded w2 (1 output/edge)
        if (t < TILE_E) {
            const int e = e0 + t;
            if (e < E) {
                float acc = bias2;
                const _Float16* row = y2 + t * SY;
#pragma unroll
                for (int k = 0; k < 128; k += 8) {
                    v8h yv = *(const v8h*)(row + k);
                    v8h wv = *(const v8h*)(lds_w2 + k);
#pragma unroll
                    for (int i = 0; i < 8; ++i)
                        acc += (float)yv[i] * (float)wv[i];
                }
                out[e] = acc;
            }
        }
        __syncthreads();   // y2 (lds_x) must be read before next gather
    }
}

// --------------------------- launcher --------------------------------------
extern "C" void kernel_launch(void* const* d_in, const int* in_sizes, int n_in,
                              void* d_out, int out_size, void* d_ws, size_t ws_size,
                              hipStream_t stream) {
    const float* h   = (const float*)d_in[0];
    const int*   src = (const int*)  d_in[1];
    const int*   dst = (const int*)  d_in[2];
    const float* W0  = (const float*)d_in[3];
    const float* b0  = (const float*)d_in[4];
    const float* W1  = (const float*)d_in[5];
    const float* b1  = (const float*)d_in[6];
    const float* W2  = (const float*)d_in[7];
    const float* b2  = (const float*)d_in[8];
    const float* g0  = (const float*)d_in[9];
    const float* be0 = (const float*)d_in[10];
    const float* g1  = (const float*)d_in[11];
    const float* be1 = (const float*)d_in[12];
    const float* g2  = (const float*)d_in[13];
    const float* be2 = (const float*)d_in[14];
    const float* rm0 = (const float*)d_in[15];
    const float* rv0 = (const float*)d_in[16];
    const float* rm1 = (const float*)d_in[17];
    const float* rv1 = (const float*)d_in[18];
    const float* rm2 = (const float*)d_in[19];
    const float* rv2 = (const float*)d_in[20];

    const int N = in_sizes[0] / 128;
    const int E = in_sizes[1];

    // Workspace layout (256B-aligned slabs), ~2.7 MB total.
    char* ws = (char*)d_ws;
    size_t off = 0;
    _Float16* hh  = (_Float16*)(ws + off); off += (size_t)N * 128 * 2;
    off = (off + 255) & ~(size_t)255;
    _Float16* W0f = (_Float16*)(ws + off); off += 128 * 256 * 2;
    _Float16* W1f = (_Float16*)(ws + off); off += 128 * 128 * 2;
    _Float16* w2f = (_Float16*)(ws + off); off += 256;
    float*    b0f = (float*)(ws + off);    off += 512;
    float*    b1f = (float*)(ws + off);    off += 512;
    float*    b2f = (float*)(ws + off);    off += 256;

    hipLaunchKernelGGL(fold_kernel, dim3(1), dim3(256), 0, stream,
                       W0, b0, W1, b1, W2, b2,
                       g0, be0, g1, be1, g2, be2,
                       rm0, rv0, rm1, rv1, rm2, rv2,
                       W0f, W1f, w2f, b0f, b1f, b2f);

    const int total = N * 128;
    hipLaunchKernelGGL(convert_h_kernel, dim3((total + 255) / 256), dim3(256),
                       0, stream, h, hh, total);

    const int num_tiles = (E + TILE_E - 1) / TILE_E;
    int num_blocks = num_tiles < 1250 ? num_tiles : 1250;  // ~4 tiles/block
    hipLaunchKernelGGL(edge_mlp_kernel, dim3(num_blocks), dim3(256), 0, stream,
                       hh, src, dst, W0f, b0f, W1f, b1f, w2f, b2f,
                       (float*)d_out, E, num_tiles);
}
